// Attention_54786602828307
// MI455X (gfx1250) — matrix operand order repaired and checked by reading, not compiled, not run on hardware
//
#include <hip/hip_runtime.h>
#include <hip/hip_bf16.h>
#include <stdint.h>

#define B_    64
#define S_    2048
#define INP_  512
#define HID_  512
#define NCH   8             // S-chunks per batch -> 512 workgroups
#define SCH   (S_ / NCH)    // 256 positions per chunk
#define STILES (SCH / 16)   // 16 s-tiles of width 16
#define BSTRIDE 260         // LDS row stride in dwords (4-dword pad)

typedef __attribute__((ext_vector_type(16))) __bf16 v16bf;
typedef __attribute__((ext_vector_type(8)))  float  v8f;

union FragBF  { uint32_t u[8]; uint4 q[2]; v16bf v; };
union Cn4     { uint4 q; float2 f2[2]; };

__device__ __forceinline__ uint32_t f2bf(float x) {
    uint32_t u = __float_as_uint(x);
    return (u + 0x7FFFu + ((u >> 16) & 1u)) >> 16;   // round-to-nearest-even
}
__device__ __forceinline__ uint32_t pack2bf(float lo, float hi) {
    return f2bf(lo) | (f2bf(hi) << 16);
}

// CDNA5 hardware tanh (TRANS unit). Two v_nops cover the ISA §7.4 TRANS
// result-use hazard since the compiler can't schedule into opaque asm.
__device__ __forceinline__ float hw_tanh(float x) {
    float r;
    asm volatile("v_tanh_f32 %0, %1\n\tv_nop\n\tv_nop" : "=v"(r) : "v"(x));
    return r;
}

// ---------------------------------------------------------------- Wc -> bf16
__global__ void k_cvt_wc(const float* __restrict__ Wc, uint2* __restrict__ wcb) {
    int i = blockIdx.x * blockDim.x + threadIdx.x;          // 65536 float4s
    float4 f = reinterpret_cast<const float4*>(Wc)[i];
    uint2 o; o.x = pack2bf(f.x, f.y); o.y = pack2bf(f.z, f.w);
    wcb[i] = o;
}

// ---------------------------------------------------------------- hid = inp @ Wl^T + bl
__global__ void k_hid(const float* __restrict__ inp, const float* __restrict__ Wl,
                      const float* __restrict__ bl, float* __restrict__ hid) {
    __shared__ __align__(16) float is[INP_];
    const int b = blockIdx.x, tid = threadIdx.x;
    is[tid]       = inp[b * INP_ + tid];
    is[tid + 256] = inp[b * INP_ + tid + 256];
    __syncthreads();
    #pragma unroll
    for (int hh = 0; hh < 2; ++hh) {
        const int h = tid + hh * 256;
        const float4* w = reinterpret_cast<const float4*>(Wl + (size_t)h * INP_);
        const float4* x = reinterpret_cast<const float4*>(is);
        float s = 0.f;
        #pragma unroll 4
        for (int i = 0; i < INP_ / 4; ++i) {
            float4 a = w[i], c = x[i];
            s = fmaf(a.x, c.x, fmaf(a.y, c.y, fmaf(a.z, c.z, fmaf(a.w, c.w, s))));
        }
        hid[b * HID_ + h] = s + bl[h];
    }
}

// ---------------------------------------------------------------- fused conv1x1 + tanh-attn + online softmax
__global__ __launch_bounds__(256) void k_attn(
    const float* __restrict__ context, const uint8_t* __restrict__ mask,
    const uint32_t* __restrict__ wcb, const float* __restrict__ hidws,
    const float* __restrict__ bc, const float* __restrict__ V,
    float* __restrict__ att_raw,                 // alpha region of d_out
    float* __restrict__ pm, float* __restrict__ pl, float* __restrict__ pacc)
{
    // double-buffered bf16 B tiles (ping-pong hides global latency in-wave)
    __shared__ __align__(16) uint32_t Bb[2][16 * BSTRIDE];
    // per-h constants: {V[h], hid[b][h]+bc[h]}  (bc folded out of the loop)
    __shared__ __align__(16) float2 cn_s[HID_];
    __shared__ float att_l[16], w_l[16];
    __shared__ float r_l;

    const int tid  = threadIdx.x;
    const int lane = tid & 31, wave = tid >> 5;
    const int ln   = lane & 15, g = lane >> 4;
    const int b    = blockIdx.x / NCH, ch = blockIdx.x % NCH;

    #pragma unroll
    for (int hh = 0; hh < 2; ++hh) {
        const int h = tid + hh * 256;
        cn_s[h] = make_float2(V[h], hidws[b * HID_ + h] + bc[h]);
    }

    float acc[4][8], ctxr[4][8];
    #pragma unroll
    for (int t = 0; t < 4; ++t)
        #pragma unroll
        for (int v = 0; v < 8; ++v) acc[t][v] = 0.f;

    float m_run = -INFINITY, l_run = 0.f;      // live in thread 0 only

    const float* cbase = context + ((size_t)b * S_ + (size_t)ch * SCH) * INP_;
    float4 stg[8];                             // in-flight staging registers

    // ---- staging helpers (8 float4 per thread = one 16x512 f32 tile) ----
    auto issue_stage = [&](int st1) {
        const float* tb = cbase + (size_t)st1 * 16 * INP_;
        if (st1 + 1 < STILES)   // walk HBM->L2 one more tile ahead
            __builtin_prefetch(reinterpret_cast<const char*>(tb) + 16 * INP_ * 4 + tid * 128, 0, 0);
        #pragma unroll
        for (int j = 0; j < 8; ++j) {
            int idx = j * 256 + tid;
            int row = idx >> 7, c4 = idx & 127;
            stg[j] = reinterpret_cast<const float4*>(tb + (size_t)row * INP_)[c4];
        }
    };
    auto commit_stage = [&](uint32_t* dst) {
        #pragma unroll
        for (int j = 0; j < 8; ++j) {
            int idx = j * 256 + tid;
            int row = idx >> 7, c4 = idx & 127;
            uint2 o; o.x = pack2bf(stg[j].x, stg[j].y); o.y = pack2bf(stg[j].z, stg[j].w);
            *reinterpret_cast<uint2*>(&dst[row * BSTRIDE + (c4 << 1)]) = o;
        }
    };

    // prologue: tile 0 into buffer 0
    issue_stage(0);
    commit_stage(Bb[0]);
    if (tid < 16) att_l[tid] = 0.f;
    __syncthreads();

    int p = 0;
    for (int st = 0; st < STILES; ++st) {
        if (st + 1 < STILES) issue_stage(st + 1);   // loads in flight across GEMM

        float attp0 = 0.f, attp1 = 0.f;
        #pragma unroll
        for (int t = 0; t < 4; ++t) {
            const int h0 = wave * 64 + t * 16;
            const uint32_t* arow = wcb + (size_t)(h0 + ln) * (INP_ / 2);
            const uint32_t* brow = &Bb[p][ln * BSTRIDE + 4 * g];
            v8f c = {0.f, 0.f, 0.f, 0.f, 0.f, 0.f, 0.f, 0.f};
            FragBF a0, b0;
            a0.q[0] = *reinterpret_cast<const uint4*>(arow + 4 * g);
            a0.q[1] = *reinterpret_cast<const uint4*>(arow + 8 + 4 * g);
            b0.q[0] = *reinterpret_cast<const uint4*>(brow);
            b0.q[1] = *reinterpret_cast<const uint4*>(brow + 8);
            #pragma unroll
            for (int k = 0; k < INP_; k += 32) {    // depth-1 rolling pipeline
                FragBF a1, b1;
                if (k + 32 < INP_) {
                    const int kd = (k + 32) >> 1;
                    a1.q[0] = *reinterpret_cast<const uint4*>(arow + kd + 4 * g);
                    a1.q[1] = *reinterpret_cast<const uint4*>(arow + kd + 8 + 4 * g);
                    b1.q[0] = *reinterpret_cast<const uint4*>(brow + kd);
                    b1.q[1] = *reinterpret_cast<const uint4*>(brow + kd + 8);
                }
                c = __builtin_amdgcn_wmma_f32_16x16x32_bf16(false, a0.v, false, b0.v,
                                                            (short)0, c, false, false);
                if (k + 32 < INP_) { a0 = a1; b0 = b1; }
            }
            // batched constant fetch: 8 contiguous float2 (h0+8g .. h0+8g+7) = 4x b128
            Cn4 cn[4];
            const uint4* cp = reinterpret_cast<const uint4*>(&cn_s[h0 + 8 * g]);
            #pragma unroll
            for (int q = 0; q < 4; ++q) cn[q].q = cp[q];
            #pragma unroll
            for (int v = 0; v < 8; ++v) {
                float2 vc = cn[v >> 1].f2[v & 1];           // {V[h], hid+bc[h]}
                ctxr[t][v] = c[v];                          // raw ctx (bc folded out)
                float pr = vc.x * hw_tanh(c[v] + vc.y);
                if (v & 1) attp1 += pr; else attp0 += pr;
            }
        }
        float attp = attp0 + attp1;
        attp += __shfl_xor(attp, 16);
        if (lane < 16) atomicAdd(&att_l[ln], attp);
        __syncthreads();

        if (tid == 0) {                                     // online softmax step
            const int sg0 = ch * SCH + st * 16;
            float av[16], mt = -INFINITY;
            #pragma unroll
            for (int j = 0; j < 16; ++j) {
                float raw = att_l[j];
                bool mk = mask[(size_t)b * S_ + sg0 + j] != 0;
                av[j] = mk ? -INFINITY : raw;
                att_raw[(size_t)b * S_ + sg0 + j] = av[j];
                mt = fmaxf(mt, av[j]);
            }
            float m_new = fmaxf(m_run, mt);
            float r, sw = 0.f;
            if (m_new == -INFINITY) {
                r = 1.f;
                #pragma unroll
                for (int j = 0; j < 16; ++j) w_l[j] = 0.f;
            } else {
                r = __expf(m_run - m_new);
                #pragma unroll
                for (int j = 0; j < 16; ++j) { float w = __expf(av[j] - m_new); w_l[j] = w; sw += w; }
            }
            l_run = l_run * r + sw; m_run = m_new; r_l = r;
        }
        __syncthreads();

        {   // rescale + accumulate raw weighted context (registers only)
            float r = r_l, w = w_l[ln];
            #pragma unroll
            for (int t = 0; t < 4; ++t)
                #pragma unroll
                for (int v = 0; v < 8; ++v)
                    acc[t][v] = acc[t][v] * r + w * ctxr[t][v];
        }
        if (tid < 16) att_l[tid] = 0.f;                     // for next tile
        if (st + 1 < STILES) commit_stage(Bb[p ^ 1]);       // convert + ds_store
        __syncthreads();                                    // Bb[p^1] & att_l ready
        p ^= 1;
    }

    // reduce acc over the 16 s-lanes of each half-wave
    #pragma unroll
    for (int t = 0; t < 4; ++t)
        #pragma unroll
        for (int v = 0; v < 8; ++v) {
            float a = acc[t][v];
            a += __shfl_xor(a, 1);
            a += __shfl_xor(a, 2);
            a += __shfl_xor(a, 4);
            a += __shfl_xor(a, 8);
            acc[t][v] = a;
        }
    if (ln == 0) {
        float* dst = pacc + (size_t)(b * NCH + ch) * HID_;
        #pragma unroll
        for (int t = 0; t < 4; ++t)
            #pragma unroll
            for (int v = 0; v < 8; ++v)
                dst[wave * 64 + t * 16 + v + 8 * g] = acc[t][v];
    }
    if (tid == 0) { pm[b * NCH + ch] = m_run; pl[b * NCH + ch] = l_run; }
}

// ---------------------------------------------------------------- merge chunk partials (+bc, folded out of main loop)
__global__ void k_comb(const float* __restrict__ pm, const float* __restrict__ pl,
                       const float* __restrict__ pacc, const float* __restrict__ bc,
                       float* __restrict__ out_hid,
                       float* __restrict__ Mb, float* __restrict__ Lb) {
    const int b = blockIdx.x, h = threadIdx.x;
    float M = -INFINITY;
    #pragma unroll
    for (int c = 0; c < NCH; ++c) M = fmaxf(M, pm[b * NCH + c]);
    float L = 0.f, a = 0.f;
    #pragma unroll
    for (int c = 0; c < NCH; ++c) {
        float e = __expf(pm[b * NCH + c] - M);
        L += pl[b * NCH + c] * e;
        a += e * pacc[(size_t)(b * NCH + c) * HID_ + h];
    }
    out_hid[b * HID_ + h] = a / L + bc[h];   // sum(alpha)=1 => +bc exactly
    if (h == 0) { Mb[b] = M; Lb[b] = L; }
}

// ---------------------------------------------------------------- normalize alpha in place
__global__ void k_alpha(float* __restrict__ alpha, const float* __restrict__ Mb,
                        const float* __restrict__ Lb) {
    int i = blockIdx.x * blockDim.x + threadIdx.x;
    int b = i >> 11;                                        // S_ == 2048
    alpha[i] = __expf(alpha[i] - Mb[b]) / Lb[b];
}

// ----------------------------------------------------------------
extern "C" void kernel_launch(void* const* d_in, const int* in_sizes, int n_in,
                              void* d_out, int out_size, void* d_ws, size_t ws_size,
                              hipStream_t stream) {
    const float*   inp  = (const float*)d_in[0];
    const float*   ctx  = (const float*)d_in[1];
    const uint8_t* mask = (const uint8_t*)d_in[2];
    const float*   Wl   = (const float*)d_in[3];
    const float*   bl   = (const float*)d_in[4];
    const float*   Wc   = (const float*)d_in[5];
    const float*   bc   = (const float*)d_in[6];
    const float*   V    = (const float*)d_in[7];

    float* out       = (float*)d_out;
    float* out_hid   = out;                    // [B, HID]
    float* out_alpha = out + B_ * HID_;        // [B, S]

    char* ws = (char*)d_ws;                    // ~1.63 MB used
    uint32_t* wcb = (uint32_t*)(ws);           // 524288 B : Wc bf16
    float* hid  = (float*)(ws + 524288);       // 131072 B
    float* pm   = (float*)(ws + 655360);       //   2048 B
    float* pl   = (float*)(ws + 657408);       //   2048 B
    float* pacc = (float*)(ws + 659456);       // 1048576 B
    float* Mb   = (float*)(ws + 1708032);      //    256 B
    float* Lb   = (float*)(ws + 1708288);      //    256 B

    k_cvt_wc<<<256, 256, 0, stream>>>(Wc, (uint2*)wcb);
    k_hid  <<<B_, 256, 0, stream>>>(inp, Wl, bl, hid);
    k_attn <<<B_ * NCH, 256, 0, stream>>>(ctx, mask, wcb, hid, bc, V,
                                          out_alpha, pm, pl, pacc);
    k_comb <<<B_, HID_, 0, stream>>>(pm, pl, pacc, bc, out_hid, Mb, Lb);
    k_alpha<<<(B_ * S_) / 256, 256, 0, stream>>>(out_alpha, Mb, Lb);
}
